// MoELayer_18476949307966
// MI455X (gfx1250) — compile-verified, matrix operand order checked by reading, not run on hardware
//
#include <hip/hip_runtime.h>
#include <hip/hip_bf16.h>

#define D_DIM 1024
#define E_EXP 8
#define T_TOK 8192
#define BM 64            // tokens per block tile
#define KC 32            // k-chunk (matches bf16 WMMA K)
#define THREADS 512      // 16 waves (wave32) -> 4 waves/SIMD, full 256-VGPR budget
#define MAX_TILES 128    // ceil(T_TOK / BM)

#define XH_STR (D_DIM + 8)       // padded row stride (2064B: 16B-aligned, conflict-free)
#define WS_STR (KC + 8)          // padded W row stride (80B: 16B-aligned, conflict-free)
#define WBUF   (D_DIM * WS_STR)  // elements per W chunk buffer

typedef __bf16 bf16;
typedef __attribute__((ext_vector_type(16))) __bf16 v16bf;
typedef __attribute__((ext_vector_type(8)))  float  v8f;

union FragB { v16bf v; unsigned int u[8]; };

// K-pair base for 16-bit A/B WMMA fragments (ISA 7.12.2):
// VGPR r in 0..3 -> K = 2r (+8 for hi lane half); r in 4..7 -> K = 16+2(r-4) (+8 hi half)
__device__ __forceinline__ int kmap(int r, int hi) {
  return ((r & 3) << 1) + ((r >> 2) << 4) + (hi << 3);
}

__device__ __forceinline__ float gelu_tanh(float x) {
  const float c0 = 0.7978845608028654f;  // sqrt(2/pi)
  float x3 = x * x * x;
  return 0.5f * x * (1.0f + tanhf(c0 * (x + 0.044715f * x3)));
}

// ---------------- init: zero combined output + expert counters ----------------
__global__ void moe_init(float* __restrict__ out_combined, int* __restrict__ counts) {
  size_t tid = (size_t)blockIdx.x * blockDim.x + threadIdx.x;
  size_t base = tid * 8;
  if (base < (size_t)T_TOK * D_DIM) {
    float4 z = {0.f, 0.f, 0.f, 0.f};
    *(float4*)(out_combined + base) = z;
    *(float4*)(out_combined + base + 4) = z;
  }
  if (blockIdx.x == 0 && threadIdx.x < E_EXP) counts[threadIdx.x] = 0;
}

// ---------------- router: logits, softmax, top-2, expert lists ----------------
__global__ __launch_bounds__(256) void moe_router(
    const float* __restrict__ x, const float* __restrict__ wr,
    float* __restrict__ logits_out, int* __restrict__ counts,
    int* __restrict__ tlist, float* __restrict__ plist) {
  __shared__ float sWr[D_DIM * E_EXP];  // 32 KB
  int tid = threadIdx.x;
  for (int i = tid; i < D_DIM * E_EXP; i += 256) sWr[i] = wr[i];
  __syncthreads();

  int t = blockIdx.x * 256 + tid;
  const float* xr = x + (size_t)t * D_DIM;
  float acc[E_EXP] = {};
  for (int d = 0; d < D_DIM; d += 4) {
    float4 xv = *(const float4*)(xr + d);
#pragma unroll
    for (int e = 0; e < E_EXP; ++e) {
      acc[e] += xv.x * sWr[(d + 0) * E_EXP + e] + xv.y * sWr[(d + 1) * E_EXP + e]
              + xv.z * sWr[(d + 2) * E_EXP + e] + xv.w * sWr[(d + 3) * E_EXP + e];
    }
  }
  // TEMP = 1.0 : logits == acc
  float mx = acc[0];
#pragma unroll
  for (int e = 1; e < E_EXP; ++e) mx = fmaxf(mx, acc[e]);
  float p[E_EXP]; float sum = 0.f;
#pragma unroll
  for (int e = 0; e < E_EXP; ++e) { p[e] = __expf(acc[e] - mx); sum += p[e]; }
  float inv = 1.f / sum;
#pragma unroll
  for (int e = 0; e < E_EXP; ++e) {
    p[e] *= inv;
    logits_out[(size_t)t * E_EXP + e] = acc[e];
  }
  // top-2 (ties -> lower index, matching lax.top_k)
  int i0 = 0;
#pragma unroll
  for (int e = 1; e < E_EXP; ++e) if (p[e] > p[i0]) i0 = e;
  int i1 = (i0 == 0) ? 1 : 0;
#pragma unroll
  for (int e = 0; e < E_EXP; ++e) if (e != i0 && p[e] > p[i1]) i1 = e;

  int pos0 = atomicAdd(&counts[i0], 1);
  tlist[i0 * T_TOK + pos0] = t; plist[i0 * T_TOK + pos0] = p[i0];
  int pos1 = atomicAdd(&counts[i1], 1);
  tlist[i1 * T_TOK + pos1] = t; plist[i1 * T_TOK + pos1] = p[i1];
}

// ---------------- fused expert FFN (grouped GEMM, bf16 WMMA) ----------------

// Load one [KC x D] f32 weight chunk, convert, and store transposed bf16 into LDS.
// Per iteration a thread handles columns (n, n+1) of k-rows (k, k+1):
//   two coalesced global_load_b64, two v_cvt_pk_bf16_f32, two ds_store_b32.
// Short-lived registers: nothing held across the WMMA section (spill-safe).
__device__ __forceinline__ void wchunk_load_cvt_store(
    const float* __restrict__ wsrc, bf16* __restrict__ dst, int tid) {
#pragma unroll
  for (int i = 0; i < KC / 2; ++i) {          // 16 k-row pairs
    int k = i * 2;
    int n = tid * 2;                          // 512 threads cover 1024 columns
    float2 r0 = *(const float2*)(wsrc + (size_t)k * D_DIM + n);
    float2 r1 = *(const float2*)(wsrc + (size_t)(k + 1) * D_DIM + n);
    // pack (k, k+1) pairs for columns n and n+1
    bf16 p0lo = (bf16)r0.x, p0hi = (bf16)r1.x;
    bf16 p1lo = (bf16)r0.y, p1hi = (bf16)r1.y;
    bf16* d0 = dst + (size_t)(n + 0) * WS_STR + k;
    bf16* d1 = dst + (size_t)(n + 1) * WS_STR + k;
    d0[0] = p0lo; d0[1] = p0hi;
    d1[0] = p1lo; d1[1] = p1hi;
  }
}

// One GEMM: acc[2][8] (32 rows x 128 cols per wave) = sA[BM x D] @ gW[D x D],
// f32 weights streamed through a double-buffered transposed bf16 LDS tile.
__device__ __forceinline__ void gemm_tile(
    const bf16* __restrict__ sA, const float* __restrict__ gW,
    bf16* __restrict__ sW, v8f (&acc)[2][8],
    int tid, int ln, int hi, int mbase, int wn0) {
  v8f z = {0.f, 0.f, 0.f, 0.f, 0.f, 0.f, 0.f, 0.f};
#pragma unroll
  for (int mt = 0; mt < 2; ++mt)
#pragma unroll
    for (int nt = 0; nt < 8; ++nt) acc[mt][nt] = z;

  // prologue: chunk 0 -> buffer 0 (and prefetch chunk 1)
  __builtin_prefetch(gW + (size_t)KC * D_DIM + tid * 64, 0, 1);
  wchunk_load_cvt_store(gW, sW, tid);
  __syncthreads();

  int buf = 0;
  for (int kc = 0; kc < D_DIM / KC; ++kc) {
    // 1) stage NEXT chunk into the other buffer (loads hit cache thanks to prefetch)
    if (kc + 1 < D_DIM / KC) {
      if (kc + 2 < D_DIM / KC)
        __builtin_prefetch(gW + (size_t)(kc + 2) * KC * D_DIM + tid * 64, 0, 1);
      wchunk_load_cvt_store(gW + (size_t)(kc + 1) * KC * D_DIM,
                            sW + (buf ^ 1) * WBUF, tid);
    }
    // 2) compute on the CURRENT chunk
    const bf16* wb = sW + buf * WBUF;
    FragB a[2];
#pragma unroll
    for (int mt = 0; mt < 2; ++mt) {
      const bf16* src = sA + (size_t)(mbase + mt * 16 + ln) * XH_STR + kc * KC;
#pragma unroll
      for (int r = 0; r < 8; ++r)
        a[mt].u[r] = *(const unsigned int*)(src + kmap(r, hi));
    }
#pragma unroll
    for (int nt = 0; nt < 8; ++nt) {
      FragB b;
      const bf16* src = wb + (size_t)(wn0 + nt * 16 + ln) * WS_STR;
#pragma unroll
      for (int r = 0; r < 8; ++r)
        b.u[r] = *(const unsigned int*)(src + kmap(r, hi));
#pragma unroll
      for (int mt = 0; mt < 2; ++mt) {
        acc[mt][nt] = __builtin_amdgcn_wmma_f32_16x16x32_bf16(
            false, a[mt].v, false, b.v, (short)0, acc[mt][nt], false, false);
      }
    }
    // 3) single barrier per chunk: stores to buf^1 done, reads of buf done
    __syncthreads();
    buf ^= 1;
  }
}

__global__ __launch_bounds__(THREADS, 1) void moe_ffn(
    const float* __restrict__ x, const float* __restrict__ w1,
    const float* __restrict__ b1, const float* __restrict__ w2,
    const float* __restrict__ b2, const int* __restrict__ counts,
    const int* __restrict__ tlist, const float* __restrict__ plist,
    float* __restrict__ out) {
  // sXH holds X (bf16) during GEMM1; X is dead once H is written, so H aliases it.
  __shared__ bf16 sXH[BM * XH_STR];     // 64*1032*2 = 129 KB
  __shared__ bf16 sW[2 * WBUF];         // 2*1024*40*2 = 160 KB (double-buffered)
  __shared__ int   sT[BM];
  __shared__ float sP[BM];

  const int e = blockIdx.y;
  const int tile = blockIdx.x;
  const int cnt = counts[e];
  if (tile * BM >= cnt) return;     // uniform early-exit before any barrier

  const int tid  = threadIdx.x;
  const int wave = tid >> 5;
  const int lane = tid & 31;
  const int ln   = lane & 15;
  const int hi   = lane >> 4;
  const int wn0  = (wave & 7) * 128;   // 8 n-slices of 128 cols
  const int mbase = (wave >> 3) * 32;  // 2 m-groups of 32 rows

  if (tid < BM) {
    int idx = tile * BM + tid;
    if (idx < cnt) { sT[tid] = tlist[e * T_TOK + idx]; sP[tid] = plist[e * T_TOK + idx]; }
    else           { sT[tid] = 0;                      sP[tid] = 0.f; }
  }
  __syncthreads();

  // gather BM token rows of x -> bf16 LDS (8 threads per row, coalesced 32-float spans)
  {
    int row = tid >> 3;
    int l8  = tid & 7;
    const float* xr = x + (size_t)sT[row] * D_DIM;
    for (int j = 0; j < D_DIM / 32; ++j) {
      int c = j * 32 + l8 * 4;
      float4 v = *(const float4*)(xr + c);
      bf16* d = sXH + (size_t)row * XH_STR + c;
      d[0] = (bf16)v.x; d[1] = (bf16)v.y; d[2] = (bf16)v.z; d[3] = (bf16)v.w;
    }
  }
  // (first barrier inside gemm_tile also covers the gather)

  v8f acc[2][8];

  // GEMM1: H = gelu(X @ W1[e] + b1[e]); all reads of sXH complete at gemm_tile's
  // final barrier, so writing H back into sXH below is race-free.
  gemm_tile(sXH, w1 + (size_t)e * D_DIM * D_DIM, sW, acc, tid, ln, hi, mbase, wn0);
#pragma unroll
  for (int mt = 0; mt < 2; ++mt)
#pragma unroll
    for (int nt = 0; nt < 8; ++nt) {
      int col = wn0 + nt * 16 + ln;
      float bias = b1[e * D_DIM + col];
#pragma unroll
      for (int r = 0; r < 8; ++r) {
        int row = mbase + mt * 16 + hi * 8 + r;  // C/D layout: VGPR r -> M=r (+8 hi half)
        sXH[(size_t)row * XH_STR + col] = (bf16)gelu_tanh(acc[mt][nt][r] + bias);
      }
    }
  __syncthreads();

  // GEMM2: out_tile = H @ W2[e] + b2[e]
  gemm_tile(sXH, w2 + (size_t)e * D_DIM * D_DIM, sW, acc, tid, ln, hi, mbase, wn0);

  // scale by routing prob and combine into the shared output
#pragma unroll
  for (int mt = 0; mt < 2; ++mt)
#pragma unroll
    for (int nt = 0; nt < 8; ++nt) {
      int col = wn0 + nt * 16 + ln;
      float bias = b2[e * D_DIM + col];
#pragma unroll
      for (int r = 0; r < 8; ++r) {
        int row = mbase + mt * 16 + hi * 8 + r;
        float v = (acc[mt][nt][r] + bias) * sP[row];
        atomicAdd(out + (size_t)sT[row] * D_DIM + col, v);
      }
    }
}

// ---------------- host launcher ----------------
extern "C" void kernel_launch(void* const* d_in, const int* in_sizes, int n_in,
                              void* d_out, int out_size, void* d_ws, size_t ws_size,
                              hipStream_t stream) {
  const float* x  = (const float*)d_in[0];
  const float* wr = (const float*)d_in[1];
  const float* w1 = (const float*)d_in[2];
  const float* b1 = (const float*)d_in[3];
  const float* w2 = (const float*)d_in[4];
  const float* b2 = (const float*)d_in[5];

  float* out = (float*)d_out;                       // combined: [T, D]
  float* logits = out + (size_t)T_TOK * D_DIM;      // logits:   [T, E]

  int*   counts = (int*)d_ws;                       // [E]
  int*   tlist  = counts + 16;                      // [E, T]
  float* plist  = (float*)(tlist + E_EXP * T_TOK);  // [E, T]

  moe_init<<<dim3((T_TOK * D_DIM / 8) / 256), dim3(256), 0, stream>>>(out, counts);
  moe_router<<<dim3(T_TOK / 256), dim3(256), 0, stream>>>(x, wr, logits, counts, tlist, plist);
  moe_ffn<<<dim3(MAX_TILES, E_EXP), dim3(THREADS), 0, stream>>>(
      x, w1, b1, w2, b2, counts, tlist, plist, out);
}
